// SpeechT5Attention_884763263358
// MI455X (gfx1250) — matrix operand order repaired and checked by reading, not compiled, not run on hardware
//
#include <hip/hip_runtime.h>

typedef __bf16 bf16;
typedef __attribute__((ext_vector_type(16))) __bf16 bf16x16;
typedef __attribute__((ext_vector_type(8)))  __bf16 bf16x8;
typedef __attribute__((ext_vector_type(8)))  float  f32x8;
typedef __attribute__((ext_vector_type(4)))  float  f32x4;

#define B_   4
#define T_   1024
#define D_   768
#define H_   12
#define HD_  64
#define BH_  (B_ * H_)

static __device__ __forceinline__ f32x8 wmma_bf16(bf16x16 a, bf16x16 b, f32x8 c) {
  return __builtin_amdgcn_wmma_f32_16x16x32_bf16(false, a, false, b, (short)0, c, false, false);
}

// A fragment (16x32 bf16, MxK). lane: m = lane%16, h = lane/16.
// elements 0..7  : K = kb + 8h + (0..7)
// elements 8..15 : K = kb + 16 + 8h + (0..7)
static __device__ __forceinline__ bf16x16 a_frag_f32(const float* base, int ld, int m, int h, int kb) {
  const float* p = base + (size_t)m * ld + kb + h * 8;
  f32x4 x0 = *(const f32x4*)(p);
  f32x4 x1 = *(const f32x4*)(p + 4);
  f32x4 y0 = *(const f32x4*)(p + 16);
  f32x4 y1 = *(const f32x4*)(p + 20);
  bf16x16 r;
#pragma unroll
  for (int i = 0; i < 4; ++i) {
    r[i]      = (bf16)x0[i];
    r[4 + i]  = (bf16)x1[i];
    r[8 + i]  = (bf16)y0[i];
    r[12 + i] = (bf16)y1[i];
  }
  return r;
}

static __device__ __forceinline__ bf16x16 a_frag_bf16(const bf16* base, int ld, int m, int h, int kb) {
  const bf16* p = base + (size_t)m * ld + kb + h * 8;
  bf16x8 lo = *(const bf16x8*)(p);
  bf16x8 hi = *(const bf16x8*)(p + 16);
  bf16x16 r;
#pragma unroll
  for (int i = 0; i < 8; ++i) { r[i] = lo[i]; r[8 + i] = hi[i]; }
  return r;
}

// B fragment (32x16 bf16, KxN). lane: n = lane%16, h = lane/16; elements 0..7: K = kb + 8h + (0..7), 8..15: K = kb + 16 + 8h + (0..7).
// Source stores column n as a K-contiguous row n (i.e. B^T row-major).
static __device__ __forceinline__ bf16x16 b_frag_bf16(const bf16* base, int ld, int n, int h, int kb) {
  const bf16* p = base + (size_t)n * ld + kb + h * 8;
  bf16x8 lo = *(const bf16x8*)(p);
  bf16x8 hi = *(const bf16x8*)(p + 16);
  bf16x16 r;
#pragma unroll
  for (int i = 0; i < 8; ++i) { r[i] = lo[i]; r[8 + i] = hi[i]; }
  return r;
}

// ---------------- Kernel 0: one-shot f32 -> bf16 conversion -----------------------
__global__ __launch_bounds__(256) void k_cvt(const float* __restrict__ src,
                                             bf16* __restrict__ dst, int n8) {
  int i = blockIdx.x * 256 + threadIdx.x;
  if (i < n8) {
    const f32x4* p = (const f32x4*)src + (size_t)i * 2;
    f32x4 a = p[0], b = p[1];
    bf16x8 o;
#pragma unroll
    for (int j = 0; j < 4; ++j) { o[j] = (bf16)a[j]; o[4 + j] = (bf16)b[j]; }
    ((bf16x8*)dst)[i] = o;
  }
}

// ---------------- Kernel 1: QKV projection (x @ W^T + b), all-bf16 GEMM -----------
// grid: (256 m-tiles, 12 heads, 3 {q,k,v}); 1 wave computes a 16x64 tile, K=768.
__global__ __launch_bounds__(32) void k_qkv(
    const bf16* __restrict__ xb,                                  // [4096][768]
    const bf16* __restrict__ Wqb, const float* __restrict__ bq,
    const bf16* __restrict__ Wkb, const float* __restrict__ bk,
    const bf16* __restrict__ Wvb, const float* __restrict__ bv,
    bf16* __restrict__ qh,   // [BH][T][HD]
    bf16* __restrict__ q2,   // [T][BH][HD]
    bf16* __restrict__ kh,   // [BH][T][HD]
    bf16* __restrict__ vT)   // [BH][HD][T]
{
  const int lane = threadIdx.x & 31;
  const int l16 = lane & 15, h = lane >> 4;
  const int m0   = blockIdx.x * 16;
  const int head = blockIdx.y;
  const int z    = blockIdx.z;        // 0:q 1:k 2:v
  const bf16*  W  = (z == 0) ? Wqb : (z == 1) ? Wkb : Wvb;
  const float* bb = (z == 0) ? bq  : (z == 1) ? bk  : bv;
  const int n0 = head * HD_;

  f32x8 acc[4];
#pragma unroll
  for (int j = 0; j < 4; ++j)
#pragma unroll
    for (int r = 0; r < 8; ++r) acc[j][r] = 0.f;

  for (int k0 = 0; k0 < D_; k0 += 32) {
    bf16x16 a = a_frag_bf16(xb + (size_t)m0 * D_, D_, l16, h, k0);
#pragma unroll
    for (int j = 0; j < 4; ++j) {
      bf16x16 b = b_frag_bf16(W + (size_t)(n0 + j * 16) * D_, D_, l16, h, k0);
      acc[j] = wmma_bf16(a, b, acc[j]);
    }
  }

  const int bidx = m0 >> 10;
  const int tb   = m0 & (T_ - 1);
  const int bh   = bidx * H_ + head;
#pragma unroll
  for (int j = 0; j < 4; ++j) {
    const int d = j * 16 + l16;
    const float beta = bb[n0 + d];
    if (z == 2) {
      bf16x8 pk;                      // 8 t-contiguous values -> one 16B store
#pragma unroll
      for (int r = 0; r < 8; ++r) pk[r] = (bf16)(acc[j][r] + beta);
      *(bf16x8*)(vT + ((size_t)bh * HD_ + d) * T_ + tb + 8 * h) = pk;
    } else {
#pragma unroll
      for (int r = 0; r < 8; ++r) {
        const int t = tb + 8 * h + r;
        float val = acc[j][r] + beta;
        if (z == 0) {
          val *= 0.125f;              // hd^-0.5
          bf16 q16 = (bf16)val;
          qh[((size_t)bh * T_ + t) * HD_ + d]  = q16;
          q2[((size_t)t * BH_ + bh) * HD_ + d] = q16;
        } else {
          kh[((size_t)bh * T_ + t) * HD_ + d] = (bf16)val;
        }
      }
    }
  }
}

// ------------- Kernel 2: relative-position bias, batched over (b,h) ---------------
// For fixed t: (pb[t] : S x 64) @ (q2[t]^T : 64 x 48) -> bias[s][bh], a true GEMM.
// All 3 bh-tiles computed per workgroup so pb streams through HBM exactly once.
// grid: (s-tiles=64 on x for q2-row L2 reuse, t=1024 on y).
__global__ __launch_bounds__(32) void k_bias(
    const float* __restrict__ pb,   // [T][T][HD]
    const bf16* __restrict__ q2,    // [T][BH][HD]
    bf16* __restrict__ bias_ws)     // [BH][T][T]
{
  const int lane = threadIdx.x & 31;
  const int l16 = lane & 15, h = lane >> 4;
  const int s0 = blockIdx.x * 16;
  const int t  = blockIdx.y;

  const float* pbt = pb + (size_t)t * T_ * HD_;
  const bf16*  qt  = q2 + (size_t)t * BH_ * HD_;

  f32x8 acc[3];
#pragma unroll
  for (int z = 0; z < 3; ++z)
#pragma unroll
    for (int r = 0; r < 8; ++r) acc[z][r] = 0.f;

#pragma unroll
  for (int k0 = 0; k0 < HD_; k0 += 32) {
    bf16x16 a = a_frag_f32(pbt + (size_t)s0 * HD_, HD_, l16, h, k0);   // rows = s
#pragma unroll
    for (int z = 0; z < 3; ++z) {
      bf16x16 b = b_frag_bf16(qt + (size_t)(z * 16) * HD_, HD_, l16, h, k0); // cols = bh
      acc[z] = wmma_bf16(a, b, acc[z]);
    }
  }
  // D element r: s = s0 + 8h + r (contiguous), bh = z*16 + l16 -> one 16B store each.
#pragma unroll
  for (int z = 0; z < 3; ++z) {
    bf16x8 pk;
#pragma unroll
    for (int r = 0; r < 8; ++r) pk[r] = (bf16)acc[z][r];
    *(bf16x8*)(bias_ws + ((size_t)(z * 16 + l16) * T_ + t) * T_ + s0 + 8 * h) = pk;
  }
}

// ------------- Kernel 3: attention with bias-seeded scores + softmax --------------
// grid: (t-tiles=64, heads=12, batch=4); 4 waves cooperate on 16 t-rows.
#define SCLD 1032   // padded row stride (word stride %64 != 0 -> conflict-light)

__global__ __launch_bounds__(128) void k_attn(
    const bf16* __restrict__ qh, const bf16* __restrict__ kh,
    const bf16* __restrict__ vT, const bf16* __restrict__ bias_ws,
    bf16* __restrict__ ctx)   // [B][T][D]
{
  __shared__ float sc[16 * SCLD];      // 66.0 KB  scores (f32)
  __shared__ bf16  pbuf[16 * SCLD];    // 33.0 KB  probabilities (bf16)
  __shared__ float rsum[16];

  const int lane = threadIdx.x & 31;
  const int wv   = threadIdx.x >> 5;   // 0..3
  const int l16 = lane & 15, h = lane >> 4;
  const int t0   = blockIdx.x * 16;
  const int head = blockIdx.y;
  const int b    = blockIdx.z;
  const int bh   = b * H_ + head;

  const bf16* q  = qh + ((size_t)bh * T_ + t0) * HD_;
  const bf16* k  = kh + (size_t)bh * T_ * HD_;
  const bf16* v  = vT + (size_t)bh * HD_ * T_;
  const bf16* bw = bias_ws + ((size_t)bh * T_ + t0) * T_;

  bf16x16 qa0 = a_frag_bf16(q, HD_, l16, h, 0);
  bf16x16 qa1 = a_frag_bf16(q, HD_, l16, h, 32);

  // Pass A: scores = q @ k^T + bias -> LDS; waves split the s-tiles.
  for (int s0 = wv * 16; s0 < T_; s0 += 64) {
    f32x8 c;
#pragma unroll
    for (int r = 0; r < 8; ++r)
      c[r] = (float)bw[(size_t)(8 * h + r) * T_ + s0 + l16];
    bf16x16 kb0 = b_frag_bf16(k + (size_t)s0 * HD_, HD_, l16, h, 0);
    bf16x16 kb1 = b_frag_bf16(k + (size_t)s0 * HD_, HD_, l16, h, 32);
    c = wmma_bf16(qa0, kb0, c);
    c = wmma_bf16(qa1, kb1, c);
#pragma unroll
    for (int r = 0; r < 8; ++r)
      sc[(8 * h + r) * SCLD + s0 + l16] = c[r];
  }
  __syncthreads();

  // Pass B: row softmax; wave wv owns rows [4*wv, 4*wv+4), lanes sweep a row
  // cooperatively (conflict-free consecutive addresses), wave-wide reductions.
  for (int t = wv * 4; t < wv * 4 + 4; ++t) {
    const float* row = sc + t * SCLD;
    bf16* prow = pbuf + t * SCLD;
    float mx = -3.0e38f;
    for (int s = lane; s < T_; s += 32) mx = fmaxf(mx, row[s]);
#pragma unroll
    for (int o = 16; o >= 1; o >>= 1) mx = fmaxf(mx, __shfl_xor(mx, o, 32));
    float sum = 0.f;
    for (int s = lane; s < T_; s += 32) {
      float e = __expf(row[s] - mx);
      sum += e;
      prow[s] = (bf16)e;
    }
#pragma unroll
    for (int o = 16; o >= 1; o >>= 1) sum += __shfl_xor(sum, o, 32);
    if (lane == 0) rsum[t] = 1.0f / sum;
  }
  __syncthreads();

  // Pass C: out = P @ V; wave wv owns d-column block j = wv (16 columns).
  f32x8 o;
#pragma unroll
  for (int r = 0; r < 8; ++r) o[r] = 0.f;

  for (int s0 = 0; s0 < T_; s0 += 32) {
    bf16x16 pa = a_frag_bf16(pbuf, SCLD, l16, h, s0);
    bf16x16 vb = b_frag_bf16(v, T_, wv * 16 + l16, h, s0);
    o = wmma_bf16(pa, vb, o);
  }
  const int d = head * HD_ + wv * 16 + l16;
#pragma unroll
  for (int r = 0; r < 8; ++r) {
    const int t = t0 + 8 * h + r;
    ctx[((size_t)b * T_ + t) * D_ + d] = (bf16)(o[r] * rsum[8 * h + r]);
  }
}

// ---------------- Kernel 4: output projection (ctx @ Wo^T + bo) -> f32 ------------
__global__ __launch_bounds__(32) void k_oproj(
    const bf16* __restrict__ ctx, const bf16* __restrict__ Wob,
    const float* __restrict__ bo, float* __restrict__ out)
{
  const int lane = threadIdx.x & 31;
  const int l16 = lane & 15, h = lane >> 4;
  const int m0 = blockIdx.x * 16;
  const int n0 = blockIdx.y * 64;

  f32x8 acc[4];
#pragma unroll
  for (int j = 0; j < 4; ++j)
#pragma unroll
    for (int r = 0; r < 8; ++r) acc[j][r] = 0.f;

  for (int k0 = 0; k0 < D_; k0 += 32) {
    bf16x16 a = a_frag_bf16(ctx + (size_t)m0 * D_, D_, l16, h, k0);
#pragma unroll
    for (int j = 0; j < 4; ++j) {
      bf16x16 bfr = b_frag_bf16(Wob + (size_t)(n0 + j * 16) * D_, D_, l16, h, k0);
      acc[j] = wmma_bf16(a, bfr, acc[j]);
    }
  }
#pragma unroll
  for (int j = 0; j < 4; ++j) {
    const int n = n0 + j * 16 + l16;
    const float beta = bo[n];
#pragma unroll
    for (int r = 0; r < 8; ++r)
      out[(size_t)(m0 + 8 * h + r) * D_ + n] = acc[j][r] + beta;
  }
}

extern "C" void kernel_launch(void* const* d_in, const int* in_sizes, int n_in,
                              void* d_out, int out_size, void* d_ws, size_t ws_size,
                              hipStream_t stream) {
  const float* x  = (const float*)d_in[0];
  const float* pb = (const float*)d_in[1];
  const float* Wq = (const float*)d_in[2];
  const float* bq = (const float*)d_in[3];
  const float* Wk = (const float*)d_in[4];
  const float* bk = (const float*)d_in[5];
  const float* Wv = (const float*)d_in[6];
  const float* bv = (const float*)d_in[7];
  const float* Wo = (const float*)d_in[8];
  const float* bo = (const float*)d_in[9];
  float* out = (float*)d_out;

  char* w = (char*)d_ws;
  bf16* qh  = (bf16*)w; w += (size_t)BH_ * T_ * HD_ * 2;   // 6.29 MB
  bf16* kh  = (bf16*)w; w += (size_t)BH_ * T_ * HD_ * 2;   // 6.29 MB
  bf16* vT  = (bf16*)w; w += (size_t)BH_ * T_ * HD_ * 2;   // 6.29 MB
  bf16* q2  = (bf16*)w; w += (size_t)T_ * BH_ * HD_ * 2;   // 6.29 MB
  bf16* cx  = (bf16*)w; w += (size_t)B_ * T_ * D_ * 2;     // 6.29 MB
  bf16* xb  = (bf16*)w; w += (size_t)B_ * T_ * D_ * 2;     // 6.29 MB
  bf16* Wqb = (bf16*)w; w += (size_t)D_ * D_ * 2;          // 1.18 MB
  bf16* Wkb = (bf16*)w; w += (size_t)D_ * D_ * 2;
  bf16* Wvb = (bf16*)w; w += (size_t)D_ * D_ * 2;
  bf16* Wob = (bf16*)w; w += (size_t)D_ * D_ * 2;
  bf16* bws = (bf16*)w;                                    // 96 MB: [BH][T][T]

  const int nX = B_ * T_ * D_ / 8;   // 393216
  const int nW = D_ * D_ / 8;        // 73728
  k_cvt<<<dim3((nX + 255) / 256), 256, 0, stream>>>(x,  xb,  nX);
  k_cvt<<<dim3((nW + 255) / 256), 256, 0, stream>>>(Wq, Wqb, nW);
  k_cvt<<<dim3((nW + 255) / 256), 256, 0, stream>>>(Wk, Wkb, nW);
  k_cvt<<<dim3((nW + 255) / 256), 256, 0, stream>>>(Wv, Wvb, nW);
  k_cvt<<<dim3((nW + 255) / 256), 256, 0, stream>>>(Wo, Wob, nW);

  k_qkv  <<<dim3(256, 12, 3), 32, 0, stream>>>(xb, Wqb, bq, Wkb, bk, Wvb, bv, qh, q2, kh, vT);
  k_bias <<<dim3(64, 1024), 32, 0, stream>>>(pb, q2, bws);
  k_attn <<<dim3(64, 12, 4), 128, 0, stream>>>(qh, kh, vT, bws, cx);
  k_oproj<<<dim3(256, 12, 1), 32, 0, stream>>>(cx, Wob, bo, out);

  (void)in_sizes; (void)n_in; (void)out_size; (void)ws_size;
}
